// CharRNNModel_14104672600793
// MI455X (gfx1250) — compile-verified, matrix operand order checked by reading, not run on hardware
//
#include <hip/hip_runtime.h>
#include <hip/hip_bf16.h>
#include <math.h>

// ---- problem dims ----
constexpr int CB  = 64;    // batch
constexpr int CS  = 256;   // seq len
constexpr int CH  = 1024;  // hidden
constexpr int CE  = 512;   // embed
constexpr int CV  = 65;    // vocab
constexpr int CH2 = 512;   // hidden/2
constexpr float CEPS = 1e-8f;

typedef __attribute__((ext_vector_type(16))) __bf16 v16bf;
typedef __attribute__((ext_vector_type(8)))  float  v8f;

// Load a 16-element bf16 WMMA fragment for one lane.
// CDNA5 16-bit A/B layout: lane L<16 holds K [kb..kb+7] and [kb+16..kb+23]
// (kb = ktile*32), lane L>=16 holds the +8 shifted chunks. Both chunks are
// 16B contiguous -> two b128 loads.
__device__ __forceinline__ v16bf load_frag(const __bf16* p) {
  union { v16bf v; uint4 q[2]; } u;
  u.q[0] = *reinterpret_cast<const uint4*>(p);
  u.q[1] = *reinterpret_cast<const uint4*>(p + 16);
  return u.v;
}

__device__ __forceinline__ v8f wmma_bf16(v16bf a, v16bf b, v8f c) {
  return __builtin_amdgcn_wmma_f32_16x16x32_bf16(
      /*neg_a=*/false, a, /*neg_b=*/false, b,
      /*c_mod=*/(short)0, c, /*reuse_a=*/false, /*reuse_b=*/false);
}

__device__ __forceinline__ float sigmoidf_(float x) {
  return 1.0f / (1.0f + expf(-x));
}

// ------------------------------------------------------------------
// Prologue kernels
// ------------------------------------------------------------------

// P0[v][h] = embedding[v] . i2h_w0[h] + i2h_b0[h]   (65 x 1024)
__global__ __launch_bounds__(256) void k_p0(const float* __restrict__ emb,
                                            const float* __restrict__ i2h_w,
                                            const float* __restrict__ i2h_b,
                                            float* __restrict__ P0) {
  int idx = blockIdx.x * 256 + threadIdx.x;
  if (idx >= CV * CH) return;
  int v = idx / CH, h = idx % CH;
  const float* e = emb + (size_t)v * CE;
  const float* w = i2h_w + (size_t)h * CE;
  float s = i2h_b[h];
  for (int k = 0; k < CE; ++k) s += e[k] * w[k];
  P0[idx] = s;
}

__global__ __launch_bounds__(256) void k_f2bf(__bf16* __restrict__ dst,
                                              const float* __restrict__ src,
                                              int n) {
  int i = blockIdx.x * 256 + threadIdx.x;
  if (i < n) dst[i] = (__bf16)src[i];
}

__global__ __launch_bounds__(256) void k_copy(float* __restrict__ dst,
                                              const float* __restrict__ src,
                                              int n) {
  int i = blockIdx.x * 256 + threadIdx.x;
  if (i < n) dst[i] = src[i];
}

// ------------------------------------------------------------------
// Recurrent step kernels (WMMA). Block = 256 threads = 8 waves.
// Wave w: m-tile = w&3 (M=64 -> 4 tiles), n-tile = (w>>2) within a 32-col
// block chunk. Each wave owns the PAIRED tiles (n, n+512) so the XOR gate
// can be formed in registers.
// ------------------------------------------------------------------

// Cell part 1: hcand = P0[token] + h_prev @ h2h_wT + h2h_b ; hn = sstan(tanh)
// also emits xor_out (bf16) for columns < 512.
__global__ __launch_bounds__(256) void k_a0(const __bf16* __restrict__ hbf,
                                            const __bf16* __restrict__ w_h2h,
                                            const float* __restrict__ P0,
                                            const int* __restrict__ x, int t,
                                            const float* __restrict__ h2h_b,
                                            const float* __restrict__ stan,
                                            float* __restrict__ hn,
                                            __bf16* __restrict__ xorbf) {
  const int tid  = threadIdx.x;
  const int lane = tid & 31;
  const int w    = tid >> 5;
  const int mtile = w & 3;
  const int n_lo  = blockIdx.x * 32 + (w >> 2) * 16;
  const int n_hi  = n_lo + CH2;
  const int l15   = lane & 15;
  const int khalf = (lane & 16) ? 8 : 0;

  v8f acc_lo = {}; v8f acc_hi = {};
  const __bf16* aB  = hbf   + (size_t)(mtile * 16 + l15) * CH + khalf;
  const __bf16* bBl = w_h2h + (size_t)(n_lo + l15) * CH + khalf;
  const __bf16* bBh = w_h2h + (size_t)(n_hi + l15) * CH + khalf;
  for (int kt = 0; kt < CH / 32; ++kt) {
    v16bf a   = load_frag(aB  + kt * 32);
    v16bf blo = load_frag(bBl + kt * 32);
    v16bf bhi = load_frag(bBh + kt * 32);
    acc_lo = wmma_bf16(a, blo, acc_lo);
    acc_hi = wmma_bf16(a, bhi, acc_hi);
  }
  const float s1 = 1.0f + stan[0];
  const int gn_lo = n_lo + l15, gn_hi = n_hi + l15;
  const int mofs = (lane & 16) ? 8 : 0;
  for (int r = 0; r < 8; ++r) {
    int gm = mtile * 16 + r + mofs;
    int tk = x[gm * CS + t];
    float pl = P0[(size_t)tk * CH + gn_lo] + h2h_b[gn_lo];
    float ph = P0[(size_t)tk * CH + gn_hi] + h2h_b[gn_hi];
    float hl = tanhf(acc_lo[r] + pl) * s1;
    float hh = tanhf(acc_hi[r] + ph) * s1;
    hn[(size_t)gm * CH + gn_lo] = hl;
    hn[(size_t)gm * CH + gn_hi] = hh;
    float a_ = sigmoidf_(hl), b_ = sigmoidf_(hh);
    float xo = 0.5f * (a_ + b_ - 2.0f * a_ * b_);
    xorbf[(size_t)gm * CH2 + gn_lo] = (__bf16)xo;
  }
}

// Cell1 part 1: hcand = h0 @ i2h_w1T + i2h_b1 + h1_prev @ h2h_w1T + h2h_b1
__global__ __launch_bounds__(256) void k_a1(const __bf16* __restrict__ h0bf,
                                            const __bf16* __restrict__ h1bf,
                                            const __bf16* __restrict__ w_i2h,
                                            const __bf16* __restrict__ w_h2h,
                                            const float* __restrict__ i2h_b,
                                            const float* __restrict__ h2h_b,
                                            const float* __restrict__ stan,
                                            float* __restrict__ hn,
                                            __bf16* __restrict__ xorbf) {
  const int tid  = threadIdx.x;
  const int lane = tid & 31;
  const int w    = tid >> 5;
  const int mtile = w & 3;
  const int n_lo  = blockIdx.x * 32 + (w >> 2) * 16;
  const int n_hi  = n_lo + CH2;
  const int l15   = lane & 15;
  const int khalf = (lane & 16) ? 8 : 0;

  v8f acc_lo = {}; v8f acc_hi = {};
  const __bf16* a0B = h0bf  + (size_t)(mtile * 16 + l15) * CH + khalf;
  const __bf16* a1B = h1bf  + (size_t)(mtile * 16 + l15) * CH + khalf;
  const __bf16* iBl = w_i2h + (size_t)(n_lo + l15) * CH + khalf;
  const __bf16* iBh = w_i2h + (size_t)(n_hi + l15) * CH + khalf;
  const __bf16* hBl = w_h2h + (size_t)(n_lo + l15) * CH + khalf;
  const __bf16* hBh = w_h2h + (size_t)(n_hi + l15) * CH + khalf;
  for (int kt = 0; kt < CH / 32; ++kt) {
    v16bf a0 = load_frag(a0B + kt * 32);
    v16bf a1 = load_frag(a1B + kt * 32);
    acc_lo = wmma_bf16(a0, load_frag(iBl + kt * 32), acc_lo);
    acc_lo = wmma_bf16(a1, load_frag(hBl + kt * 32), acc_lo);
    acc_hi = wmma_bf16(a0, load_frag(iBh + kt * 32), acc_hi);
    acc_hi = wmma_bf16(a1, load_frag(hBh + kt * 32), acc_hi);
  }
  const float s1 = 1.0f + stan[0];
  const int gn_lo = n_lo + l15, gn_hi = n_hi + l15;
  const int mofs = (lane & 16) ? 8 : 0;
  for (int r = 0; r < 8; ++r) {
    int gm = mtile * 16 + r + mofs;
    float pl = i2h_b[gn_lo] + h2h_b[gn_lo];
    float ph = i2h_b[gn_hi] + h2h_b[gn_hi];
    float hl = tanhf(acc_lo[r] + pl) * s1;
    float hh = tanhf(acc_hi[r] + ph) * s1;
    hn[(size_t)gm * CH + gn_lo] = hl;
    hn[(size_t)gm * CH + gn_hi] = hh;
    float a_ = sigmoidf_(hl), b_ = sigmoidf_(hh);
    float xo = 0.5f * (a_ + b_ - 2.0f * a_ * b_);
    xorbf[(size_t)gm * CH2 + gn_lo] = (__bf16)xo;
  }
}

// Cell part 2: h = hn + xor_out @ xor_wT + xor_b ; emit fp32 + bf16 copies,
// optionally append into the [B,S,H] history (cell1).
__global__ __launch_bounds__(256) void k_bx(const __bf16* __restrict__ xbf,
                                            const __bf16* __restrict__ w_xor,
                                            const float* __restrict__ hn,
                                            const float* __restrict__ xor_b,
                                            float* __restrict__ hf,
                                            __bf16* __restrict__ hbf,
                                            float* __restrict__ hist, int t) {
  const int tid  = threadIdx.x;
  const int lane = tid & 31;
  const int w    = tid >> 5;
  const int mtile = w & 3;
  const int n     = blockIdx.x * 32 + (w >> 2) * 16;
  const int l15   = lane & 15;
  const int khalf = (lane & 16) ? 8 : 0;

  v8f acc = {};
  const __bf16* aB = xbf   + (size_t)(mtile * 16 + l15) * CH2 + khalf;
  const __bf16* bB = w_xor + (size_t)(n + l15) * CH2 + khalf;
  for (int kt = 0; kt < CH2 / 32; ++kt)
    acc = wmma_bf16(load_frag(aB + kt * 32), load_frag(bB + kt * 32), acc);

  const int gn = n + l15;
  const int mofs = (lane & 16) ? 8 : 0;
  for (int r = 0; r < 8; ++r) {
    int gm = mtile * 16 + r + mofs;
    float v = acc[r] + hn[(size_t)gm * CH + gn] + xor_b[gn];
    hf[(size_t)gm * CH + gn] = v;
    hbf[(size_t)gm * CH + gn] = (__bf16)v;
    if (hist) hist[(size_t)gm * CS * CH + (size_t)t * CH + gn] = v;
  }
}

// ------------------------------------------------------------------
// Head kernels (deterministic fixed-order reductions, no atomics)
// ------------------------------------------------------------------

__global__ __launch_bounds__(256) void k_var(const float* __restrict__ outs,
                                             float* __restrict__ ivar) {
  int col = blockIdx.x, tid = threadIdx.x;
  float s = 0.f, sq = 0.f;
  for (int r = tid; r < CB * CS; r += 256) {
    float v = outs[(size_t)r * CH + col];
    s += v; sq += v * v;
  }
  __shared__ float ls[256], lq[256];
  ls[tid] = s; lq[tid] = sq; __syncthreads();
  for (int st = 128; st > 0; st >>= 1) {
    if (tid < st) { ls[tid] += ls[tid + st]; lq[tid] += lq[tid + st]; }
    __syncthreads();
  }
  if (tid == 0) {
    const float N = (float)(CB * CS);
    float var = (lq[0] - ls[0] * ls[0] / N) / (N - 1.0f);  // ddof=1
    ivar[col] = 1.0f / (var + CEPS);
  }
}

__global__ __launch_bounds__(256) void k_wsq(const float* __restrict__ unemb,
                                             const float* __restrict__ ivar,
                                             float* __restrict__ wsq) {
  int v = blockIdx.x, tid = threadIdx.x;
  float s = 0.f;
  for (int h = tid; h < CH; h += 256) {
    float u = unemb[(size_t)h * CV + v];
    s += u * u * ivar[h];
  }
  __shared__ float red[256];
  red[tid] = s; __syncthreads();
  for (int st = 128; st > 0; st >>= 1) {
    if (tid < st) red[tid] += red[tid + st];
    __syncthreads();
  }
  if (tid == 0) wsq[v] = red[0];
}

__global__ __launch_bounds__(128) void k_mahal(const float* __restrict__ outs,
                                               const float* __restrict__ ivar,
                                               const float* __restrict__ unemb,
                                               const float* __restrict__ wsq,
                                               float* __restrict__ mah,
                                               float* __restrict__ parts) {
  int row = blockIdx.x, tid = threadIdx.x;
  __shared__ float so[CH];
  __shared__ float red[128];
  float osq = 0.f;
  for (int h = tid; h < CH; h += 128) {
    float v  = outs[(size_t)row * CH + h];
    float iv = ivar[h];
    so[h] = v * iv;
    osq += v * v * iv;
  }
  red[tid] = osq; __syncthreads();
  for (int st = 64; st > 0; st >>= 1) {
    if (tid < st) red[tid] += red[tid + st];
    __syncthreads();
  }
  float osq_t = red[0];
  float m = 0.f;
  if (tid < CV) {
    float cross = 0.f;
    for (int h = 0; h < CH; ++h) cross += so[h] * unemb[(size_t)h * CV + tid];
    float d2 = fmaxf(osq_t + wsq[tid] - 2.0f * cross, 0.0f);
    m = sqrtf(d2);
    mah[(size_t)row * CV + tid] = m;
  }
  __syncthreads();
  red[tid] = m; __syncthreads();
  for (int st = 64; st > 0; st >>= 1) {
    if (tid < st) red[tid] += red[tid + st];
    __syncthreads();
  }
  if (tid == 0) parts[row] = red[0];
}

__global__ __launch_bounds__(256) void k_scale(const float* __restrict__ parts,
                                               float* __restrict__ scl) {
  int tid = threadIdx.x;
  float s = 0.f;
  for (int i = tid; i < CB * CS; i += 256) s += parts[i];
  __shared__ float red[256];
  red[tid] = s; __syncthreads();
  for (int st = 128; st > 0; st >>= 1) {
    if (tid < st) red[tid] += red[tid + st];
    __syncthreads();
  }
  if (tid == 0)
    scl[0] = fmaxf(red[0] / ((float)(CB * CS) * (float)CV), CEPS);
}

__global__ __launch_bounds__(128) void k_soft(const float* __restrict__ mah,
                                              const float* __restrict__ scl,
                                              float* __restrict__ p) {
  int row = blockIdx.x, tid = threadIdx.x;
  __shared__ float li[CV];
  __shared__ float lse;
  float sc = scl[0];
  if (tid < CV) {
    float m  = mah[(size_t)row * CV + tid];
    float mc = fmaxf(m / sc, 1e-6f);
    li[tid] = -32.0f * logf(mc + CEPS);   // n = sqrt(1024) = 32
  }
  __syncthreads();
  if (tid == 0) {
    float mx = li[0];
    for (int v = 1; v < CV; ++v) mx = fmaxf(mx, li[v]);
    float se = 0.f;
    for (int v = 0; v < CV; ++v) se += expf(li[v] - mx);
    lse = mx + logf(se);
  }
  __syncthreads();
  if (tid < CV) p[(size_t)row * CV + tid] = expf(li[tid] - lse);
}

// ------------------------------------------------------------------
extern "C" void kernel_launch(void* const* d_in, const int* in_sizes, int n_in,
                              void* d_out, int out_size, void* d_ws, size_t ws_size,
                              hipStream_t stream) {
  (void)in_sizes; (void)n_in; (void)out_size; (void)ws_size;
  const int*   x      = (const int*)  d_in[0];
  const float* emb    = (const float*)d_in[1];
  const float* i2h_w0 = (const float*)d_in[2];
  const float* i2h_b0 = (const float*)d_in[3];
  const float* h2h_w0 = (const float*)d_in[4];
  const float* h2h_b0 = (const float*)d_in[5];
  const float* stan0  = (const float*)d_in[6];
  const float* xor_w0 = (const float*)d_in[7];
  const float* xor_b0 = (const float*)d_in[8];
  const float* i2h_w1 = (const float*)d_in[9];
  const float* i2h_b1 = (const float*)d_in[10];
  const float* h2h_w1 = (const float*)d_in[11];
  const float* h2h_b1 = (const float*)d_in[12];
  const float* stan1  = (const float*)d_in[13];
  const float* xor_w1 = (const float*)d_in[14];
  const float* xor_b1 = (const float*)d_in[15];
  const float* unemb  = (const float*)d_in[16];

  char* wsb = (char*)d_ws;
  size_t off = 0;
  auto take = [&](size_t bytes) -> void* {
    void* p = wsb + off;
    off += (bytes + 255) & ~(size_t)255;
    return p;
  };
  float*  P0    = (float*) take(sizeof(float) * CV * CH);
  float*  hn0   = (float*) take(sizeof(float) * CB * CH);
  float*  hn1   = (float*) take(sizeof(float) * CB * CH);
  float*  h0f   = (float*) take(sizeof(float) * CB * CH);
  float*  h1f   = (float*) take(sizeof(float) * CB * CH);
  __bf16* h0bf  = (__bf16*)take(sizeof(__bf16) * CB * CH);
  __bf16* h1bf  = (__bf16*)take(sizeof(__bf16) * CB * CH);
  __bf16* xbf0  = (__bf16*)take(sizeof(__bf16) * CB * CH2);
  __bf16* xbf1  = (__bf16*)take(sizeof(__bf16) * CB * CH2);
  __bf16* wh2h0 = (__bf16*)take(sizeof(__bf16) * CH * CH);
  __bf16* wxor0 = (__bf16*)take(sizeof(__bf16) * CH * CH2);
  __bf16* wi2h1 = (__bf16*)take(sizeof(__bf16) * CH * CH);
  __bf16* wh2h1 = (__bf16*)take(sizeof(__bf16) * CH * CH);
  __bf16* wxor1 = (__bf16*)take(sizeof(__bf16) * CH * CH2);
  float*  outsH = (float*) take(sizeof(float) * (size_t)CB * CS * CH);  // 64 MB h1 history
  float*  ivar  = (float*) take(sizeof(float) * CH);
  float*  wsq   = (float*) take(sizeof(float) * CV);
  float*  mah   = (float*) take(sizeof(float) * (size_t)CB * CS * CV);
  float*  parts = (float*) take(sizeof(float) * CB * CS);
  float*  scl   = (float*) take(sizeof(float) * 1);

  // Prologue: zero hidden states, weights -> bf16, embedding projection table
  hipMemsetAsync(h0bf, 0, sizeof(__bf16) * CB * CH, stream);
  hipMemsetAsync(h1bf, 0, sizeof(__bf16) * CB * CH, stream);
  k_f2bf<<<(CH * CH  + 255) / 256, 256, 0, stream>>>(wh2h0, h2h_w0, CH * CH);
  k_f2bf<<<(CH * CH2 + 255) / 256, 256, 0, stream>>>(wxor0, xor_w0, CH * CH2);
  k_f2bf<<<(CH * CH  + 255) / 256, 256, 0, stream>>>(wi2h1, i2h_w1, CH * CH);
  k_f2bf<<<(CH * CH  + 255) / 256, 256, 0, stream>>>(wh2h1, h2h_w1, CH * CH);
  k_f2bf<<<(CH * CH2 + 255) / 256, 256, 0, stream>>>(wxor1, xor_w1, CH * CH2);
  k_p0<<<(CV * CH + 255) / 256, 256, 0, stream>>>(emb, i2h_w0, i2h_b0, P0);

  // Recurrence: stream order enforces the sequential dependency.
  for (int t = 0; t < CS; ++t) {
    k_a0<<<16, 256, 0, stream>>>(h0bf, wh2h0, P0, x, t, h2h_b0, stan0, hn0, xbf0);
    k_bx<<<32, 256, 0, stream>>>(xbf0, wxor0, hn0, xor_b0, h0f, h0bf,
                                 (float*)nullptr, 0);
    k_a1<<<16, 256, 0, stream>>>(h0bf, h1bf, wi2h1, wh2h1, i2h_b1, h2h_b1,
                                 stan1, hn1, xbf1);
    k_bx<<<32, 256, 0, stream>>>(xbf1, wxor1, hn1, xor_b1, h1f, h1bf,
                                 outsH, t);
  }

  // Head
  k_var  <<<CH,      256, 0, stream>>>(outsH, ivar);
  k_wsq  <<<CV,      256, 0, stream>>>(unemb, ivar, wsq);
  k_mahal<<<CB * CS, 128, 0, stream>>>(outsH, ivar, unemb, wsq, mah, parts);
  k_scale<<<1,       256, 0, stream>>>(parts, scl);

  float* p_out = (float*)d_out;
  k_soft<<<CB * CS, 128, 0, stream>>>(mah, scl, p_out);
  k_copy<<<(CB * CH + 255) / 256, 256, 0, stream>>>(
      p_out + (size_t)CB * CS * CV, h0f, CB * CH);
  k_copy<<<(CB * CH + 255) / 256, 256, 0, stream>>>(
      p_out + (size_t)CB * CS * CV + CB * CH, h1f, CB * CH);
}
